// BitNetBlock2b_8048768712781
// MI455X (gfx1250) — matrix operand order repaired and checked by reading, not compiled, not run on hardware
//
#include <hip/hip_runtime.h>

// ---------- types ----------
typedef __attribute__((ext_vector_type(16))) _Float16 v16h;
typedef __attribute__((ext_vector_type(8)))  float    v8f;
typedef __attribute__((ext_vector_type(4)))  unsigned int u32x4;
typedef __attribute__((ext_vector_type(2)))  unsigned int u32x2;
typedef __attribute__((ext_vector_type(4)))  int      i32x4;

union Frag16 { v16h v; _Float16 h[16]; u32x4 u[2]; };
union Half8  { u32x4 u; _Float16 h[8]; };
union Half4  { u32x2 u; _Float16 h[4]; };

#define DIMC 1024
#define FFC  4096
#define SEQC 2048
#define TOKC 4096   // 2 * 2048

// ---------- CDNA5 async global->LDS path (guarded; sync fallback) ----------
#if __has_builtin(__builtin_amdgcn_global_load_async_to_lds_b128) && \
    __has_builtin(__builtin_amdgcn_s_wait_asynccnt)
#define USE_ASYNC_LDS 1
#else
#define USE_ASYNC_LDS 0
#endif

__device__ __forceinline__ void copy16_to_lds(const _Float16* g, _Float16* l) {
#if USE_ASYNC_LDS
    void* gv = (void*)g;
    void* lv = (void*)l;
    __builtin_amdgcn_global_load_async_to_lds_b128(
        (__attribute__((address_space(1))) i32x4*)gv,
        (__attribute__((address_space(3))) i32x4*)lv, 0, 0);
#else
    *(u32x4*)l = *(const u32x4*)g;
#endif
}

__device__ __forceinline__ void wait_async_le(int n) {
#if USE_ASYNC_LDS
    if (n == 0) __builtin_amdgcn_s_wait_asynccnt(0);
    else        __builtin_amdgcn_s_wait_asynccnt(4);
#else
    (void)n;
#endif
}

// ---------- weight quantize + scale -> f16 ----------
// w' = (rint(clamp(w,-2,2)*0.75+0.5)-0.5) * s[row]
__global__ void quant_w_kernel(const float* __restrict__ w, const float* __restrict__ s,
                               _Float16* __restrict__ dst, int cols, int total4) {
    int i = blockIdx.x * blockDim.x + threadIdx.x;
    if (i >= total4) return;
    const float4 wv = ((const float4*)w)[i];
    int row = (i * 4) / cols;
    float sc = s[row];
    float a[4] = {wv.x, wv.y, wv.z, wv.w};
    Half4 o;
    #pragma unroll
    for (int e = 0; e < 4; ++e) {
        float c = fminf(fmaxf(a[e], -2.0f), 2.0f);
        float q = rintf(c * 0.75f + 0.5f) - 0.5f;
        o.h[e] = (_Float16)(q * sc);
    }
    *(u32x2*)&dst[(size_t)i * 4] = o.u;
}

// ---------- layernorm (f32 in) -> f16 out ----------
__global__ __launch_bounds__(256)
void layernorm_f16_kernel(const float* __restrict__ x, const float* __restrict__ g,
                          const float* __restrict__ bt, _Float16* __restrict__ out) {
    const int row = blockIdx.x, tid = threadIdx.x;
    const float* xr = x + (size_t)row * DIMC;
    float vals[4];
    float s1 = 0.f, s2 = 0.f;
    #pragma unroll
    for (int e = 0; e < 4; ++e) {
        float v = xr[tid + 256 * e];
        vals[e] = v; s1 += v; s2 += v * v;
    }
    #pragma unroll
    for (int m = 16; m >= 1; m >>= 1) {
        s1 += __shfl_xor(s1, m, 32);
        s2 += __shfl_xor(s2, m, 32);
    }
    __shared__ float r1[8], r2[8];
    const int wid = tid >> 5, lane = tid & 31;
    if (lane == 0) { r1[wid] = s1; r2[wid] = s2; }
    __syncthreads();
    s1 = 0.f; s2 = 0.f;
    #pragma unroll
    for (int w = 0; w < 8; ++w) { s1 += r1[w]; s2 += r2[w]; }
    const float mu = s1 * (1.0f / DIMC);
    const float var = s2 * (1.0f / DIMC) - mu * mu;
    const float rstd = rsqrtf(var + 1e-5f);
    _Float16* orow = out + (size_t)row * DIMC;
    #pragma unroll
    for (int e = 0; e < 4; ++e) {
        int c = tid + 256 * e;
        orow[c] = (_Float16)((vals[e] - mu) * rstd * g[c] + bt[c]);
    }
}

// ---------- WMMA GEMM: C[M,N] = A[M,K](f16) * W[N,K](f16)^T + bias, optional GELU / residual ----------
// Block: 256 thr (8 waves), tile 128x128, BK=32, double-buffered LDS with async staging.
// Wave grid 2(M)x4(N); wave computes 64x32 = 4x2 wmma tiles; 8 WMMAs per wave per K-step.
template<bool OUT16, bool RES, bool GELU>
__global__ __launch_bounds__(256)
void gemm_wmma_kernel(const _Float16* __restrict__ A, const _Float16* __restrict__ W,
                      const float* __restrict__ bias, const float* __restrict__ res,
                      void* __restrict__ out, int M, int N, int K) {
    __shared__ _Float16 As[2][128 * 40];
    __shared__ _Float16 Ws[2][128 * 40];
    const int tid  = threadIdx.x;
    const int wave = tid >> 5, lane = tid & 31;
    const int g = lane >> 4, ln = lane & 15;
    const int m0 = blockIdx.y * 128, n0 = blockIdx.x * 128;
    const int wm = (wave >> 2) * 64, wn = (wave & 3) * 32;

    v8f acc[4][2];
    #pragma unroll
    for (int i = 0; i < 4; ++i)
        #pragma unroll
        for (int j = 0; j < 2; ++j)
            #pragma unroll
            for (int e = 0; e < 8; ++e) acc[i][j][e] = 0.f;

    // each thread stages 2 vec16 per matrix per tile: 4 async ops per wave per matrix pair
    const int vr0 = (tid + 0)   >> 2, vc0 = ((tid + 0)   & 3) << 3;
    const int vr1 = (tid + 256) >> 2, vc1 = ((tid + 256) & 3) << 3;

    const int nkt = K >> 5;
    // prologue: stage tile 0 into buffer 0
    {
        copy16_to_lds(&A[(size_t)(m0 + vr0) * K + vc0], &As[0][vr0 * 40 + vc0]);
        copy16_to_lds(&W[(size_t)(n0 + vr0) * K + vc0], &Ws[0][vr0 * 40 + vc0]);
        copy16_to_lds(&A[(size_t)(m0 + vr1) * K + vc1], &As[0][vr1 * 40 + vc1]);
        copy16_to_lds(&W[(size_t)(n0 + vr1) * K + vc1], &Ws[0][vr1 * 40 + vc1]);
    }
    for (int kt = 0; kt < nkt; ++kt) {
        const int cur = kt & 1;
        if (kt + 1 < nkt) {   // stage next tile into the other buffer
            const int k1 = (kt + 1) << 5;
            copy16_to_lds(&A[(size_t)(m0 + vr0) * K + k1 + vc0], &As[cur ^ 1][vr0 * 40 + vc0]);
            copy16_to_lds(&W[(size_t)(n0 + vr0) * K + k1 + vc0], &Ws[cur ^ 1][vr0 * 40 + vc0]);
            copy16_to_lds(&A[(size_t)(m0 + vr1) * K + k1 + vc1], &As[cur ^ 1][vr1 * 40 + vc1]);
            copy16_to_lds(&W[(size_t)(n0 + vr1) * K + k1 + vc1], &Ws[cur ^ 1][vr1 * 40 + vc1]);
            wait_async_le(4);   // current tile's 4 ops (issued earlier, in-order) are done
        } else {
            wait_async_le(0);
        }
        __syncthreads();

        Frag16 af[4], bf[2];
        #pragma unroll
        for (int mt = 0; mt < 4; ++mt) {
            const _Float16* p = &As[cur][(wm + mt * 16 + ln) * 40];
            af[mt].u[0] = *(const u32x4*)&p[8 * g];
            af[mt].u[1] = *(const u32x4*)&p[16 + 8 * g];
        }
        #pragma unroll
        for (int nt = 0; nt < 2; ++nt) {
            const _Float16* p = &Ws[cur][(wn + nt * 16 + ln) * 40];
            bf[nt].u[0] = *(const u32x4*)&p[8 * g];
            bf[nt].u[1] = *(const u32x4*)&p[16 + 8 * g];
        }
        #pragma unroll
        for (int mt = 0; mt < 4; ++mt)
            #pragma unroll
            for (int nt = 0; nt < 2; ++nt)
                acc[mt][nt] = __builtin_amdgcn_wmma_f32_16x16x32_f16(
                    false, af[mt].v, false, bf[nt].v, (short)0, acc[mt][nt], false, false);
        __syncthreads();   // compute done before this buffer is restaged
    }
    // epilogue: C layout -> row = base + r + 8*g, col = base + ln
    #pragma unroll
    for (int nt = 0; nt < 2; ++nt) {
        const int col = n0 + wn + nt * 16 + ln;
        const float bv = bias[col];
        #pragma unroll
        for (int mt = 0; mt < 4; ++mt) {
            #pragma unroll
            for (int r = 0; r < 8; ++r) {
                const int row = m0 + wm + mt * 16 + r + 8 * g;
                float v = acc[mt][nt][r] + bv;
                if (GELU) v = 0.5f * v * (1.0f + erff(v * 0.70710678118654752f));
                if (RES)  v += res[(size_t)row * N + col];
                if (OUT16) ((_Float16*)out)[(size_t)row * N + col] = (_Float16)v;
                else       ((float*)out)[(size_t)row * N + col] = v;
            }
        }
    }
}

// ---------- Flash attention: 16 heads, d=64, seq=2048, scale=1/8 ----------
// Block: 128 thr (4 waves), M-tile 64 (wave = 16 rows), key tiles of 64.
__global__ __launch_bounds__(128)
void attention_wmma_kernel(const _Float16* __restrict__ Q, const _Float16* __restrict__ Kk,
                           const _Float16* __restrict__ V, _Float16* __restrict__ O) {
    __shared__ _Float16 Qs[64 * 72];
    __shared__ _Float16 Ks[64 * 72];
    __shared__ _Float16 Vt[64 * 72];          // transposed: [d][j]
    __shared__ _Float16 Ps[4][16 * 72];       // per-wave P staging
    const int tid  = threadIdx.x;
    const int wave = tid >> 5, lane = tid & 31;
    const int g = lane >> 4, ln = lane & 15;
    const int mt0 = blockIdx.x * 64;
    const int h = blockIdx.y, b = blockIdx.z;
    const size_t baseTok = (size_t)b * SEQC;
    const int ch0 = h * 64;

    #pragma unroll
    for (int it = 0; it < 4; ++it) {          // load Q tile 64x64
        int vi = tid + 128 * it;
        int r = vi >> 3, c = (vi & 7) << 3;
        *(u32x4*)&Qs[r * 72 + c] = *(const u32x4*)&Q[(baseTok + mt0 + r) * DIMC + ch0 + c];
    }

    float mrow[8], lrow[8];
    v8f oacc[4];
    #pragma unroll
    for (int r = 0; r < 8; ++r) { mrow[r] = -1e30f; lrow[r] = 0.f; }
    #pragma unroll
    for (int dt = 0; dt < 4; ++dt)
        #pragma unroll
        for (int e = 0; e < 8; ++e) oacc[dt][e] = 0.f;

    const float scale = 0.125f;               // 64^-0.5
    for (int kt = 0; kt < (SEQC >> 6); ++kt) {
        __syncthreads();
        const int j0 = kt << 6;
        #pragma unroll
        for (int it = 0; it < 4; ++it) {      // K via async path, V transposed into Vt
            int vi = tid + 128 * it;
            int r = vi >> 3, c = (vi & 7) << 3;
            copy16_to_lds(&Kk[(baseTok + j0 + r) * DIMC + ch0 + c], &Ks[r * 72 + c]);
            Half8 vv; vv.u = *(const u32x4*)&V[(baseTok + j0 + r) * DIMC + ch0 + c];
            #pragma unroll
            for (int e = 0; e < 8; ++e) Vt[(c + e) * 72 + r] = vv.h[e];
        }
        wait_async_le(0);
        __syncthreads();

        // S = Q K^T (wave: 16x64)
        v8f sacc[4];
        #pragma unroll
        for (int nt = 0; nt < 4; ++nt)
            #pragma unroll
            for (int e = 0; e < 8; ++e) sacc[nt][e] = 0.f;
        #pragma unroll
        for (int ks = 0; ks < 2; ++ks) {
            const int kb = ks << 5;
            Frag16 af;
            const _Float16* p = &Qs[(wave * 16 + ln) * 72 + kb];
            af.u[0] = *(const u32x4*)&p[8 * g];
            af.u[1] = *(const u32x4*)&p[16 + 8 * g];
            #pragma unroll
            for (int nt = 0; nt < 4; ++nt) {
                Frag16 bf;
                const _Float16* q = &Ks[(nt * 16 + ln) * 72 + kb];
                bf.u[0] = *(const u32x4*)&q[8 * g];
                bf.u[1] = *(const u32x4*)&q[16 + 8 * g];
                sacc[nt] = __builtin_amdgcn_wmma_f32_16x16x32_f16(
                    false, af.v, false, bf.v, (short)0, sacc[nt], false, false);
            }
        }
        #pragma unroll
        for (int nt = 0; nt < 4; ++nt)
            #pragma unroll
            for (int e = 0; e < 8; ++e) sacc[nt][e] *= scale;

        // online softmax per row (rows live across VGPR index r; cols across 16-lane half)
        #pragma unroll
        for (int r = 0; r < 8; ++r) {
            float mx = -1e30f;
            #pragma unroll
            for (int nt = 0; nt < 4; ++nt) mx = fmaxf(mx, sacc[nt][r]);
            #pragma unroll
            for (int m = 8; m >= 1; m >>= 1) mx = fmaxf(mx, __shfl_xor(mx, m, 32));
            const float mnew  = fmaxf(mrow[r], mx);
            const float alpha = __expf(mrow[r] - mnew);
            float psum = 0.f;
            #pragma unroll
            for (int nt = 0; nt < 4; ++nt) {
                float pv = __expf(sacc[nt][r] - mnew);
                sacc[nt][r] = pv;
                psum += pv;
            }
            #pragma unroll
            for (int m = 8; m >= 1; m >>= 1) psum += __shfl_xor(psum, m, 32);
            lrow[r] = lrow[r] * alpha + psum;
            mrow[r] = mnew;
            #pragma unroll
            for (int dt = 0; dt < 4; ++dt) oacc[dt][r] *= alpha;
            #pragma unroll
            for (int nt = 0; nt < 4; ++nt)
                Ps[wave][(r + 8 * g) * 72 + nt * 16 + ln] = (_Float16)sacc[nt][r];
        }

        // O += P V  (A-frag of P from LDS restaging; B-frag contiguous from Vt)
        #pragma unroll
        for (int ks = 0; ks < 2; ++ks) {
            const int kb = ks << 5;
            Frag16 af;
            const _Float16* p = &Ps[wave][ln * 72 + kb];
            af.u[0] = *(const u32x4*)&p[8 * g];
            af.u[1] = *(const u32x4*)&p[16 + 8 * g];
            #pragma unroll
            for (int dt = 0; dt < 4; ++dt) {
                Frag16 bf;
                const _Float16* q = &Vt[(dt * 16 + ln) * 72 + kb];
                bf.u[0] = *(const u32x4*)&q[8 * g];
                bf.u[1] = *(const u32x4*)&q[16 + 8 * g];
                oacc[dt] = __builtin_amdgcn_wmma_f32_16x16x32_f16(
                    false, af.v, false, bf.v, (short)0, oacc[dt], false, false);
            }
        }
    }

    #pragma unroll
    for (int dt = 0; dt < 4; ++dt) {
        const int col = ch0 + dt * 16 + ln;
        #pragma unroll
        for (int r = 0; r < 8; ++r) {
            const int row = mt0 + wave * 16 + r + 8 * g;
            O[(baseTok + row) * DIMC + col] = (_Float16)(oacc[dt][r] / lrow[r]);
        }
    }
}

// ---------- host-side orchestration ----------
extern "C" void kernel_launch(void* const* d_in, const int* in_sizes, int n_in,
                              void* d_out, int out_size, void* d_ws, size_t ws_size,
                              hipStream_t stream) {
    const float* x    = (const float*)d_in[0];
    const float* ln1g = (const float*)d_in[1];
    const float* ln1b = (const float*)d_in[2];
    const float* wq = (const float*)d_in[3];  const float* bq = (const float*)d_in[4];  const float* sq = (const float*)d_in[5];
    const float* wk = (const float*)d_in[6];  const float* bk = (const float*)d_in[7];  const float* sk = (const float*)d_in[8];
    const float* wv = (const float*)d_in[9];  const float* bv = (const float*)d_in[10]; const float* sv = (const float*)d_in[11];
    const float* wo = (const float*)d_in[12]; const float* bo = (const float*)d_in[13]; const float* so = (const float*)d_in[14];
    const float* ln2g = (const float*)d_in[15];
    const float* ln2b = (const float*)d_in[16];
    const float* w1 = (const float*)d_in[17]; const float* b1 = (const float*)d_in[18]; const float* s1 = (const float*)d_in[19];
    const float* w2 = (const float*)d_in[20]; const float* b2 = (const float*)d_in[21]; const float* s2 = (const float*)d_in[22];
    float* out = (float*)d_out;

    char* ws = (char*)d_ws;
    size_t off = 0;
    auto take = [&](size_t bytes) -> char* {
        char* p = ws + off;
        off = (off + bytes + 255) & ~(size_t)255;
        return p;
    };
    _Float16* wq16 = (_Float16*)take((size_t)DIMC * DIMC * 2);
    _Float16* wk16 = (_Float16*)take((size_t)DIMC * DIMC * 2);
    _Float16* wv16 = (_Float16*)take((size_t)DIMC * DIMC * 2);
    _Float16* wo16 = (_Float16*)take((size_t)DIMC * DIMC * 2);
    _Float16* w116 = (_Float16*)take((size_t)FFC * DIMC * 2);
    _Float16* w216 = (_Float16*)take((size_t)DIMC * FFC * 2);
    _Float16* nx16 = (_Float16*)take((size_t)TOKC * DIMC * 2);   // reused as LN2 output
    _Float16* q16  = (_Float16*)take((size_t)TOKC * DIMC * 2);
    _Float16* k16  = (_Float16*)take((size_t)TOKC * DIMC * 2);
    _Float16* v16  = (_Float16*)take((size_t)TOKC * DIMC * 2);
    _Float16* at16 = (_Float16*)take((size_t)TOKC * DIMC * 2);
    float*    xa   = (float*)   take((size_t)TOKC * DIMC * 4);
    _Float16* ff16 = q16;   // 32MB region q16..at16 is dead by the time FF1 runs

    // 1) quantize weights -> f16
    {
        int t4 = DIMC * DIMC / 4;
        int blocks = (t4 + 255) / 256;
        quant_w_kernel<<<blocks, 256, 0, stream>>>(wq, sq, wq16, DIMC, t4);
        quant_w_kernel<<<blocks, 256, 0, stream>>>(wk, sk, wk16, DIMC, t4);
        quant_w_kernel<<<blocks, 256, 0, stream>>>(wv, sv, wv16, DIMC, t4);
        quant_w_kernel<<<blocks, 256, 0, stream>>>(wo, so, wo16, DIMC, t4);
        int t4f = FFC * DIMC / 4;
        int blocksf = (t4f + 255) / 256;
        quant_w_kernel<<<blocksf, 256, 0, stream>>>(w1, s1, w116, DIMC, t4f);
        quant_w_kernel<<<blocksf, 256, 0, stream>>>(w2, s2, w216, FFC,  t4f);
    }
    // 2) LN1
    layernorm_f16_kernel<<<TOKC, 256, 0, stream>>>(x, ln1g, ln1b, nx16);
    // 3) Q, K, V projections
    {
        dim3 grid(DIMC / 128, TOKC / 128);
        gemm_wmma_kernel<true, false, false><<<grid, 256, 0, stream>>>(nx16, wq16, bq, nullptr, q16, TOKC, DIMC, DIMC);
        gemm_wmma_kernel<true, false, false><<<grid, 256, 0, stream>>>(nx16, wk16, bk, nullptr, k16, TOKC, DIMC, DIMC);
        gemm_wmma_kernel<true, false, false><<<grid, 256, 0, stream>>>(nx16, wv16, bv, nullptr, v16, TOKC, DIMC, DIMC);
    }
    // 4) attention
    attention_wmma_kernel<<<dim3(SEQC / 64, 16, 2), 128, 0, stream>>>(q16, k16, v16, at16);
    // 5) output projection + residual -> xa (f32)
    gemm_wmma_kernel<false, true, false><<<dim3(DIMC / 128, TOKC / 128), 256, 0, stream>>>(
        at16, wo16, bo, x, xa, TOKC, DIMC, DIMC);
    // 6) LN2 -> nx16
    layernorm_f16_kernel<<<TOKC, 256, 0, stream>>>(xa, ln2g, ln2b, nx16);
    // 7) FF1 + exact GELU -> ff16
    gemm_wmma_kernel<true, false, true><<<dim3(FFC / 128, TOKC / 128), 256, 0, stream>>>(
        nx16, w116, b1, nullptr, ff16, TOKC, FFC, DIMC);
    // 8) FF2 + residual -> out (f32)
    gemm_wmma_kernel<false, true, false><<<dim3(DIMC / 128, TOKC / 128), 256, 0, stream>>>(
        ff16, w216, b2, xa, out, TOKC, DIMC, FFC);
    (void)in_sizes; (void)n_in; (void)out_size; (void)ws_size;
}
